// SpatialSTEncoder_27247272526010
// MI455X (gfx1250) — matrix operand order repaired and checked by reading, not compiled, not run on hardware
//
#include <hip/hip_runtime.h>
#include <cmath>

// ---------------------------------------------------------------------------
// CDNA5 (gfx1250): all GEMMs on v_wmma_f32_16x16x32_bf16, 128-bit staged LDS.
// ---------------------------------------------------------------------------

typedef __attribute__((ext_vector_type(16))) __bf16 v16bf;
typedef __attribute__((ext_vector_type(8)))  float  v8f;

union Frag {
    v16bf v;
    uint4 q[2];
    unsigned short u[16];
};

__device__ __forceinline__ unsigned short f2bf(float f) {
    unsigned u = __float_as_uint(f);
    u += 0x7FFFu + ((u >> 16) & 1u);     // round-to-nearest-even
    return (unsigned short)(u >> 16);
}
__device__ __forceinline__ float bf2f(unsigned short h) {
    return __uint_as_float(((unsigned)h) << 16);
}
__device__ __forceinline__ float gelu_exact(float x) {
    return 0.5f * x * (1.0f + erff(x * 0.70710678118654752f));
}
// order-preserving float<->uint encoding for atomicMax on floats
__device__ __forceinline__ unsigned fenc(float f) {
    unsigned u = __float_as_uint(f);
    return (u & 0x80000000u) ? ~u : (u | 0x80000000u);
}
__device__ __forceinline__ float fdec(unsigned e) {
    unsigned u = (e & 0x80000000u) ? (e & 0x7FFFFFFFu) : ~e;
    return __uint_as_float(u);
}

__device__ __forceinline__ float block_sum_256(float v, float* red) {
    int t = threadIdx.x;
    red[t] = v; __syncthreads();
    for (int s = 128; s > 0; s >>= 1) { if (t < s) red[t] += red[t + s]; __syncthreads(); }
    float r = red[0]; __syncthreads();
    return r;
}
__device__ __forceinline__ float block_max_256(float v, float* red) {
    int t = threadIdx.x;
    red[t] = v; __syncthreads();
    for (int s = 128; s > 0; s >>= 1) { if (t < s) red[t] = fmaxf(red[t], red[t + s]); __syncthreads(); }
    float r = red[0]; __syncthreads();
    return r;
}

// ---------------------------------------------------------------------------
// Batched WMMA GEMM: C = epi(alpha * A@B + bias) [* rowscale]
//   A (bf16): elem(m,k) at A[z0*sAb0 + z1*sAb1 + m*sAm + k*sAk]
//   B       : elem(k,n) at B[z0*sBb0 + z1*sBb1 + k*sBk + n*sBn]  (BF32: f32)
//   C       : elem(m,n) at C[z0*sCb0 + z1*sCb1 + m*sCm + n]
//   MODE: 0 f32 store, 1 f32 accumulate, 2 bf16 store, 3 bf16 gelu store
// Block: 128 threads = 4 waves; block tile 64x64; wave tile 32x32 (2x2 WMMA).
// ---------------------------------------------------------------------------
#define TM 64
#define TN 64
#define TK 32
#define LPAD 8   // LDS row padding (ushorts) -> 80B row stride, 16B-aligned runs

template<int BF32, int MODE>
__global__ __launch_bounds__(128)
void gemm_wmma(const unsigned short* __restrict__ A,
               long sAm, long sAk, long sAb0, long sAb1,
               const void* __restrict__ Bptr,
               long sBk, long sBn, long sBb0, long sBb1,
               void* __restrict__ Cptr,
               long sCm, long sCb0, long sCb1,
               const float* __restrict__ bias,
               const float* __restrict__ rowscale, long sRb0, long sRb1,
               float alpha, int M, int N, int K, int BZ0)
{
    __shared__ __align__(16) unsigned short As[TM][TK + LPAD];   // [m][k]
    __shared__ __align__(16) unsigned short Bt[TN][TK + LPAD];   // [n][k] (transposed)

    const int z  = blockIdx.z;
    const int z0 = z % BZ0;
    const int z1 = z / BZ0;
    const long aBase = (long)z0 * sAb0 + (long)z1 * sAb1;
    const long bBase = (long)z0 * sBb0 + (long)z1 * sBb1;
    const long cBase = (long)z0 * sCb0 + (long)z1 * sCb1;

    const float* Bf = (const float*)Bptr;
    const unsigned short* Bh = (const unsigned short*)Bptr;

    const int tid = threadIdx.x;
    const int rowBase = blockIdx.y * TM;
    const int colBase = blockIdx.x * TN;

    const int wid  = tid >> 5;
    const int lane = tid & 31;
    const int wr   = (wid >> 1) * 32;
    const int wc   = (wid & 1) * 32;
    const int half = lane >> 4;
    const int l16  = lane & 15;

    // block-uniform fast-path eligibility
    const bool aFastB = (sAk == 1) && (rowBase + TM <= M);
    const bool bFastB = (sBn == 1) && (colBase + TN <= N);

    v8f acc[2][2] = {};

    for (int k0 = 0; k0 < K; k0 += TK) {
        const bool kFull = (k0 + TK <= K);

        // ---- stage A tile (TM x TK) ----
        if (aFastB && kFull) {
            if (k0 + 2 * TK <= K) {   // prefetch next K tile
                int c = tid, r = c >> 2, c8 = (c & 3) << 3;
                __builtin_prefetch(A + aBase + (long)(rowBase + r) * sAm + (k0 + TK) + c8, 0, 1);
            }
            #pragma unroll
            for (int j = 0; j < 2; ++j) {
                int c = tid + j * 128;            // 0..255 chunks of 8
                int r = c >> 2, c8 = (c & 3) << 3;
                const uint4* src = (const uint4*)(A + aBase + (long)(rowBase + r) * sAm + k0 + c8);
                *(uint4*)&As[r][c8] = *src;
            }
        } else {
            for (int i = tid; i < TM * TK; i += 128) {
                int r = i / TK, c = i % TK;
                int gm = rowBase + r, gk = k0 + c;
                unsigned short v = 0;
                if (gm < M && gk < K) v = A[aBase + (long)gm * sAm + (long)gk * sAk];
                As[r][c] = v;
            }
        }

        // ---- stage B tile, transposed into Bt[n][k] ----
        if (bFastB && kFull) {
            #pragma unroll
            for (int j = 0; j < 2; ++j) {
                int c  = tid + j * 128;           // 0..255
                int kr = c >> 3;                  // 0..31
                int n8 = (c & 7) << 3;            // 0,8,..,56
                long o = bBase + (long)(k0 + kr) * sBk + colBase + n8;
                unsigned short tmp[8];
                if (BF32) {
                    const float4* s = (const float4*)(Bf + o);
                    float4 f0 = s[0], f1 = s[1];
                    tmp[0] = f2bf(f0.x); tmp[1] = f2bf(f0.y);
                    tmp[2] = f2bf(f0.z); tmp[3] = f2bf(f0.w);
                    tmp[4] = f2bf(f1.x); tmp[5] = f2bf(f1.y);
                    tmp[6] = f2bf(f1.z); tmp[7] = f2bf(f1.w);
                } else {
                    union { uint4 q; unsigned short u[8]; } ld;
                    ld.q = *(const uint4*)(Bh + o);
                    #pragma unroll
                    for (int t = 0; t < 8; ++t) tmp[t] = ld.u[t];
                }
                #pragma unroll
                for (int t = 0; t < 8; ++t) Bt[n8 + t][kr] = tmp[t];
            }
        } else {
            for (int i = tid; i < TK * TN; i += 128) {
                int r = i / TN, c = i % TN;       // r = k, c = n
                int gk = k0 + r, gn = colBase + c;
                unsigned short v = 0;
                if (gk < K && gn < N) {
                    long o = bBase + (long)gk * sBk + (long)gn * sBn;
                    v = BF32 ? f2bf(Bf[o]) : Bh[o];
                }
                Bt[c][r] = v;
            }
        }
        __syncthreads();

        // ---- fragments (ISA 7.12.2): two 16B LDS loads each ----
        Frag fa[2], fb[2];
        #pragma unroll
        for (int s = 0; s < 2; ++s) {
            const int mrow = wr + s * 16 + l16;
            fa[s].q[0] = *(const uint4*)&As[mrow][half * 8];        // k = half*8 + 0..7
            fa[s].q[1] = *(const uint4*)&As[mrow][16 + half * 8];   // k = 16 + half*8 + 0..7
            const int ncol = wc + s * 16 + l16;
            fb[s].q[0] = *(const uint4*)&Bt[ncol][half * 8];        // k = half*8 + 0..7
            fb[s].q[1] = *(const uint4*)&Bt[ncol][16 + half * 8];   // k = 16 + half*8 + 0..7
        }
        #pragma unroll
        for (int i = 0; i < 2; ++i)
            #pragma unroll
            for (int j = 0; j < 2; ++j)
                acc[i][j] = __builtin_amdgcn_wmma_f32_16x16x32_bf16(
                    false, fa[i].v, false, fb[j].v, (short)0, acc[i][j], false, false);
        __syncthreads();
    }

    // ---- epilogue (C/D layout: VGPR r -> row r + 8*half, col = lane&15) ----
    #pragma unroll
    for (int i = 0; i < 2; ++i) {
        #pragma unroll
        for (int j = 0; j < 2; ++j) {
            #pragma unroll
            for (int r = 0; r < 8; ++r) {
                int m = rowBase + wr + i * 16 + half * 8 + r;
                int n = colBase + wc + j * 16 + l16;
                if (m < M && n < N) {
                    float v = alpha * acc[i][j][r];
                    if (bias)     v += bias[n];
                    if (rowscale) v *= rowscale[(long)z0 * sRb0 + (long)z1 * sRb1 + m];
                    long co = cBase + (long)m * sCm + n;
                    if (MODE == 0)      ((float*)Cptr)[co] = v;
                    else if (MODE == 1) ((float*)Cptr)[co] += v;
                    else if (MODE == 2) ((unsigned short*)Cptr)[co] = f2bf(v);
                    else                ((unsigned short*)Cptr)[co] = f2bf(gelu_exact(v));
                }
            }
        }
    }
}

// ---------------------------------------------------------------------------
// Elementwise / reduction kernels
// ---------------------------------------------------------------------------

__global__ __launch_bounds__(256)
void layernorm_rows(const float* __restrict__ x, const float* __restrict__ g,
                    const float* __restrict__ b, unsigned short* __restrict__ y)
{
    __shared__ float red[256];
    const int D = 256;
    long row = blockIdx.x;
    int t = threadIdx.x;
    float v = x[row * D + t];
    float mu = block_sum_256(v, red) * (1.0f / D);
    float d = v - mu;
    float var = block_sum_256(d * d, red) * (1.0f / D);
    float r = rsqrtf(var + 1e-5f);
    y[row * D + t] = f2bf(d * r * g[t] + b[t]);
}

__global__ __launch_bounds__(256)
void embed_tokens(const float* __restrict__ expr, const float* __restrict__ val_w,
                  const float* __restrict__ val_b, const float* __restrict__ lg,
                  const float* __restrict__ lb, const float* __restrict__ gene_emb,
                  const float* __restrict__ pos_emb, float* __restrict__ X,
                  int Kk, int Ntok)
{
    __shared__ float red[256];
    const int D = 256;
    int bk = blockIdx.x;
    int b = bk / Kk, k = bk % Kk;
    int t = threadIdx.x;
    float e = expr[(long)b * Kk + k];
    float v = e * val_w[t] + val_b[t];
    float mu = block_sum_256(v, red) * (1.0f / D);
    float d = v - mu;
    float var = block_sum_256(d * d, red) * (1.0f / D);
    float h = gelu_exact(d * rsqrtf(var + 1e-5f) * lg[t] + lb[t]);
    X[((long)b * Ntok + 1 + k) * D + t] = gene_emb[(long)k * D + t] + pos_emb[(long)k * D + t] + h;
}

__global__ __launch_bounds__(256)
void embed_spatial(const float* __restrict__ coords, const float* __restrict__ sp_w,
                   const float* __restrict__ sp_b, const float* __restrict__ lg,
                   const float* __restrict__ lb, float* __restrict__ X, int Ntok)
{
    __shared__ float red[256];
    const int D = 256;
    int b = blockIdx.x;
    int t = threadIdx.x;
    float v = coords[b * 2 + 0] * sp_w[t] + coords[b * 2 + 1] * sp_w[D + t] + sp_b[t];
    float mu = block_sum_256(v, red) * (1.0f / D);
    float d = v - mu;
    float var = block_sum_256(d * d, red) * (1.0f / D);
    X[(long)b * Ntok * D + t] = gelu_exact(d * rsqrtf(var + 1e-5f) * lg[t] + lb[t]);
}

__global__ void feat_rowstats(const unsigned short* __restrict__ Xh,
                              const unsigned short* __restrict__ XD,
                              float* __restrict__ diag, float* __restrict__ rmax,
                              unsigned* __restrict__ gmax,
                              int H, int Ntok, int D, int DH, int Mf, int Mfp,
                              float norm2, int doGlobal, long rows)
{
    long idx = (long)blockIdx.x * blockDim.x + threadIdx.x;
    if (idx >= rows) return;
    int n = (int)(idx % Ntok);
    long bh = idx / Ntok;
    int h = (int)(bh % H);
    int b = (int)(bh / H);
    const unsigned short* xr = Xh + ((long)b * Ntok + n) * D + (long)h * DH;
    float s = 0.f;
    for (int d = 0; d < DH; ++d) { float f = bf2f(xr[d]); s += f * f; }
    diag[idx] = 0.5f * s * norm2;
    const unsigned short* xd = XD + idx * Mfp;
    float mx = -3.4e38f;
    for (int m = 0; m < Mf; ++m) mx = fmaxf(mx, bf2f(xd[m]));
    rmax[idx] = mx;
    if (doGlobal) atomicMax(gmax, fenc(mx));
}

__global__ void feat_expmap(unsigned short* __restrict__ XD,
                            const float* __restrict__ diag, const float* __restrict__ rmax,
                            const unsigned* __restrict__ gmax, int useGlobal,
                            long rows, int Mf, int Mfp, float ratio, float keps)
{
    long i = (long)blockIdx.x * blockDim.x + threadIdx.x;
    long total = rows * (long)Mfp;
    if (i >= total) return;
    long r = i / Mfp;
    int m = (int)(i % Mfp);
    float out = 0.f;
    if (m < Mf) {
        float stab = useGlobal ? fdec(*gmax) : rmax[r];
        out = ratio * (expf(bf2f(XD[i]) - diag[r] - stab) + keps);
    }
    XD[i] = f2bf(out);
}

__global__ void kf_colsum(const unsigned short* __restrict__ KF, float* __restrict__ ksum,
                          int BH, int Ntok, int Mfp)
{
    int i = blockIdx.x * blockDim.x + threadIdx.x;
    if (i >= BH * Mfp) return;
    int m = i % Mfp, bh = i / Mfp;
    const unsigned short* p = KF + (long)bh * Ntok * Mfp + m;
    float s = 0.f;
    for (int n = 0; n < Ntok; ++n) s += bf2f(p[(long)n * Mfp]);
    ksum[i] = s;
}

__global__ void qf_dinv(const unsigned short* __restrict__ QF, const float* __restrict__ ksum,
                        float* __restrict__ dinv, int Ntok, int Mfp, long rows)
{
    long i = (long)blockIdx.x * blockDim.x + threadIdx.x;
    if (i >= rows) return;
    long bh = i / Ntok;
    const unsigned short* q = QF + i * (long)Mfp;
    const float* ks = ksum + bh * Mfp;
    float s = 0.f;
    for (int m = 0; m < Mfp; ++m) s += bf2f(q[m]) * ks[m];
    dinv[i] = 1.0f / s;
}

__global__ void proj_transpose(const float* __restrict__ proj, unsigned short* __restrict__ projT,
                               int Mf, int DH, int Mfp)
{
    int i = blockIdx.x * blockDim.x + threadIdx.x;
    if (i >= DH * Mfp) return;
    int k = i / Mfp, m = i % Mfp;
    projT[i] = (m < Mf) ? f2bf(proj[(long)m * DH + k]) : (unsigned short)0;
}

__global__ void cvt_f32_bf16(const float* __restrict__ in, unsigned short* __restrict__ out, long n)
{
    long i = (long)blockIdx.x * blockDim.x + threadIdx.x;
    if (i < n) out[i] = f2bf(in[i]);
}

__global__ void init_u32(unsigned* p, unsigned v) { *p = v; }

__global__ void pool_scores(const float* __restrict__ Q, const float* __restrict__ KKm,
                            float* __restrict__ sc, int Kk, int D, float invs, long total)
{
    long i = (long)blockIdx.x * blockDim.x + threadIdx.x;
    if (i >= total) return;
    long b = i / Kk;
    const float* q = Q + b * D;
    const float* kr = KKm + i * D;
    float s = 0.f;
    for (int d = 0; d < D; ++d) s += q[d] * kr[d];
    sc[i] = s * invs;
}

__global__ __launch_bounds__(256)
void pool_softmax(float* __restrict__ sc, int Kk)
{
    __shared__ float red[256];
    float* s = sc + (long)blockIdx.x * Kk;
    float mx = -3.4e38f;
    for (int i = threadIdx.x; i < Kk; i += 256) mx = fmaxf(mx, s[i]);
    mx = block_max_256(mx, red);
    float sum = 0.f;
    for (int i = threadIdx.x; i < Kk; i += 256) { float e = expf(s[i] - mx); s[i] = e; sum += e; }
    sum = block_sum_256(sum, red);
    float inv = 1.0f / sum;
    for (int i = threadIdx.x; i < Kk; i += 256) s[i] *= inv;
}

__global__ void pool_weighted(const float* __restrict__ attn, const float* __restrict__ VVm,
                              float* __restrict__ pooled, int Kk, int D, long total)
{
    long i = (long)blockIdx.x * blockDim.x + threadIdx.x;
    if (i >= total) return;
    long b = i / D;
    int d = (int)(i % D);
    const float* a = attn + b * Kk;
    const float* v = VVm + b * (long)Kk * D + d;
    float s = 0.f;
    for (int k = 0; k < Kk; ++k) s += a[k] * v[(long)k * D];
    pooled[i] = s;
}

// ---------------------------------------------------------------------------
// Host side
// ---------------------------------------------------------------------------
static void gemm(hipStream_t st,
                 const unsigned short* A, long sAm, long sAk, long sAb0, long sAb1,
                 const void* Bm, int bF32, long sBk, long sBn, long sBb0, long sBb1,
                 void* C, int mode, long sCm, long sCb0, long sCb1,
                 const float* bias, const float* rs, long sRb0, long sRb1,
                 float alpha, int M, int N, int K, int batches, int BZ0)
{
    dim3 g((N + TN - 1) / TN, (M + TM - 1) / TM, batches);
#define GEMM_CASE(BF, MD)                                                        \
    gemm_wmma<BF, MD><<<g, 128, 0, st>>>(A, sAm, sAk, sAb0, sAb1,                \
                                         Bm, sBk, sBn, sBb0, sBb1,               \
                                         C, sCm, sCb0, sCb1, bias, rs,           \
                                         sRb0, sRb1, alpha, M, N, K, BZ0)
    if (bF32) {
        if      (mode == 0) GEMM_CASE(1, 0);
        else if (mode == 1) GEMM_CASE(1, 1);
        else if (mode == 2) GEMM_CASE(1, 2);
        else                GEMM_CASE(1, 3);
    } else {
        if      (mode == 0) GEMM_CASE(0, 0);
        else if (mode == 1) GEMM_CASE(0, 1);
        else if (mode == 2) GEMM_CASE(0, 2);
        else                GEMM_CASE(0, 3);
    }
#undef GEMM_CASE
}

extern "C" void kernel_launch(void* const* d_in, const int* in_sizes, int n_in,
                              void* d_out, int out_size, void* d_ws, size_t ws_size,
                              hipStream_t stream)
{
    (void)in_sizes; (void)n_in; (void)out_size; (void)ws_size;
    constexpr int B = 32, Kk = 2048, D = 256, H = 4, DH = 64, LYR = 2;
    constexpr int NTOK = Kk + 1, FF = 4 * D, MF = 266, MFP = 288;
    const long BN = (long)B * NTOK;
    const long BH = (long)B * H;
    const float normv = 1.0f / sqrtf(sqrtf((float)DH));   // 64^-0.25
    const float norm2 = normv * normv;                    // 0.125
    const float ratio = 1.0f / sqrtf((float)MF);
    const float keps  = 1e-4f;

    const float* expr     = (const float*)d_in[0];
    const float* coords   = (const float*)d_in[1];
    const float* gene_emb = (const float*)d_in[2];
    const float* pos_emb  = (const float*)d_in[3];
    const float* val_w    = (const float*)d_in[4];
    const float* val_b    = (const float*)d_in[5];
    const float* val_ln_g = (const float*)d_in[6];
    const float* val_ln_b = (const float*)d_in[7];
    const float* sp_w     = (const float*)d_in[8];
    const float* sp_b     = (const float*)d_in[9];
    const float* sp_ln_g  = (const float*)d_in[10];
    const float* sp_ln_b  = (const float*)d_in[11];
    const float* p_ln1_g  = (const float*)d_in[12];
    const float* p_ln1_b  = (const float*)d_in[13];
    const float* p_wq     = (const float*)d_in[14];
    const float* p_bq     = (const float*)d_in[15];
    const float* p_wk     = (const float*)d_in[16];
    const float* p_bk     = (const float*)d_in[17];
    const float* p_wv     = (const float*)d_in[18];
    const float* p_bv     = (const float*)d_in[19];
    const float* p_wo     = (const float*)d_in[20];
    const float* p_bo     = (const float*)d_in[21];
    const float* p_ln2_g  = (const float*)d_in[22];
    const float* p_ln2_b  = (const float*)d_in[23];
    const float* p_ff1_w  = (const float*)d_in[24];
    const float* p_ff1_b  = (const float*)d_in[25];
    const float* p_ff2_w  = (const float*)d_in[26];
    const float* p_ff2_b  = (const float*)d_in[27];
    const float* projW    = (const float*)d_in[28];
    const float* q_w      = (const float*)d_in[29];
    const float* q_b      = (const float*)d_in[30];
    const float* k_w      = (const float*)d_in[31];
    const float* k_b      = (const float*)d_in[32];
    const float* v_w      = (const float*)d_in[33];
    const float* v_b      = (const float*)d_in[34];
    const float* o_w      = (const float*)d_in[35];
    const float* o_b      = (const float*)d_in[36];

    size_t off = 0;
    auto alloc = [&](size_t bytes) -> void* {
        void* p = (char*)d_ws + off;
        off += (bytes + 255) & ~(size_t)255;
        return p;
    };
    float*          X     = (float*)         alloc((size_t)BN * D * 4);
    unsigned short* Ybf   = (unsigned short*)alloc((size_t)BN * D * 2);
    unsigned short* Qbf   = (unsigned short*)alloc((size_t)BN * D * 2);
    unsigned short* Kbf   = (unsigned short*)alloc((size_t)BN * D * 2);
    unsigned short* Vbf   = (unsigned short*)alloc((size_t)BN * D * 2);
    unsigned short* QF    = (unsigned short*)alloc((size_t)BH * NTOK * MFP * 2);
    unsigned short* KF    = (unsigned short*)alloc((size_t)BH * NTOK * MFP * 2);
    float*          diag  = (float*)         alloc((size_t)BH * NTOK * 4);
    float*          rmax  = (float*)         alloc((size_t)BH * NTOK * 4);
    unsigned*       gmax  = (unsigned*)      alloc(256);
    float*          ksum  = (float*)         alloc((size_t)BH * MFP * 4);
    float*          dinv  = (float*)         alloc((size_t)BH * NTOK * 4);
    float*          ctx   = (float*)         alloc((size_t)BH * MFP * DH * 4);
    unsigned short* Obf   = (unsigned short*)alloc((size_t)BN * D * 2);
    unsigned short* projT = (unsigned short*)alloc((size_t)DH * MFP * 2);
    float*          Qsp   = (float*)         alloc((size_t)B * D * 4);
    float*          score = (float*)         alloc((size_t)B * Kk * 4);
    float*          pool  = (float*)         alloc((size_t)B * D * 4);
    unsigned short* poolb = (unsigned short*)alloc((size_t)B * D * 2);
    // aliases over dead regions
    unsigned short* FFH = QF;                  // BN*FF bf16 <= QF bytes
    float* KKm = (float*)KF;                   // B*Kk*D f32
    float* VVm = KKm + (size_t)B * Kk * D;     // B*Kk*D f32 (inside KF region)

    const long rowsBHN = BH * (long)NTOK;
    const long elemsF  = rowsBHN * (long)MFP;

    embed_spatial<<<B, 256, 0, stream>>>(coords, sp_w, sp_b, sp_ln_g, sp_ln_b, X, NTOK);
    embed_tokens<<<B * Kk, 256, 0, stream>>>(expr, val_w, val_b, val_ln_g, val_ln_b,
                                             gene_emb, pos_emb, X, Kk, NTOK);

    for (int l = 0; l < LYR; ++l) {
        const float* wq = p_wq + (long)l * D * D;  const float* bq = p_bq + (long)l * D;
        const float* wk = p_wk + (long)l * D * D;  const float* bk = p_bk + (long)l * D;
        const float* wv = p_wv + (long)l * D * D;  const float* bv = p_bv + (long)l * D;
        const float* wo = p_wo + (long)l * D * D;  const float* bo = p_bo + (long)l * D;

        layernorm_rows<<<(unsigned)BN, 256, 0, stream>>>(X, p_ln1_g + (long)l * D,
                                                         p_ln1_b + (long)l * D, Ybf);
        gemm(stream, Ybf, D, 1, 0, 0, wq, 1, D, 1, 0, 0, Qbf, 2, D, 0, 0,
             bq, nullptr, 0, 0, 1.0f, (int)BN, D, D, 1, 1);
        gemm(stream, Ybf, D, 1, 0, 0, wk, 1, D, 1, 0, 0, Kbf, 2, D, 0, 0,
             bk, nullptr, 0, 0, 1.0f, (int)BN, D, D, 1, 1);
        gemm(stream, Ybf, D, 1, 0, 0, wv, 1, D, 1, 0, 0, Vbf, 2, D, 0, 0,
             bv, nullptr, 0, 0, 1.0f, (int)BN, D, D, 1, 1);

        proj_transpose<<<(DH * MFP + 255) / 256, 256, 0, stream>>>(
            projW + (long)l * MF * DH, projT, MF, DH, MFP);

        // feature map q
        gemm(stream, Qbf, D, 1, DH, (long)NTOK * D,
             projT, 0, MFP, 1, 0, 0,
             QF, 2, MFP, (long)NTOK * MFP, (long)H * NTOK * MFP,
             nullptr, nullptr, 0, 0, normv, NTOK, MFP, DH, (int)BH, H);
        feat_rowstats<<<(unsigned)((rowsBHN + 255) / 256), 256, 0, stream>>>(
            Qbf, QF, diag, rmax, gmax, H, NTOK, D, DH, MF, MFP, norm2, 0, rowsBHN);
        feat_expmap<<<(unsigned)((elemsF + 255) / 256), 256, 0, stream>>>(
            QF, diag, rmax, gmax, 0, rowsBHN, MF, MFP, ratio, keps);

        // feature map k (global stabilizer)
        gemm(stream, Kbf, D, 1, DH, (long)NTOK * D,
             projT, 0, MFP, 1, 0, 0,
             KF, 2, MFP, (long)NTOK * MFP, (long)H * NTOK * MFP,
             nullptr, nullptr, 0, 0, normv, NTOK, MFP, DH, (int)BH, H);
        init_u32<<<1, 1, 0, stream>>>(gmax, 0u);
        feat_rowstats<<<(unsigned)((rowsBHN + 255) / 256), 256, 0, stream>>>(
            Kbf, KF, diag, rmax, gmax, H, NTOK, D, DH, MF, MFP, norm2, 1, rowsBHN);
        feat_expmap<<<(unsigned)((elemsF + 255) / 256), 256, 0, stream>>>(
            KF, diag, rmax, gmax, 1, rowsBHN, MF, MFP, ratio, keps);

        kf_colsum<<<(int)((BH * MFP + 255) / 256), 256, 0, stream>>>(KF, ksum, (int)BH, NTOK, MFP);
        qf_dinv<<<(unsigned)((rowsBHN + 255) / 256), 256, 0, stream>>>(QF, ksum, dinv, NTOK, MFP, rowsBHN);

        // context = kf^T @ v
        gemm(stream, KF, 1, MFP, (long)NTOK * MFP, (long)H * NTOK * MFP,
             Vbf, 0, D, 1, DH, (long)NTOK * D,
             ctx, 0, DH, (long)MFP * DH, (long)H * MFP * DH,
             nullptr, nullptr, 0, 0, 1.0f, MFP, DH, NTOK, (int)BH, H);

        // out = (qf @ context) * d_inv -> Obf
        gemm(stream, QF, MFP, 1, (long)NTOK * MFP, (long)H * NTOK * MFP,
             ctx, 1, DH, 1, (long)MFP * DH, (long)H * MFP * DH,
             Obf, 2, D, DH, (long)NTOK * D,
             nullptr, dinv, NTOK, (long)H * NTOK,
             1.0f, NTOK, DH, MFP, (int)BH, H);

        // X += Obf @ wo + bo
        gemm(stream, Obf, D, 1, 0, 0, wo, 1, D, 1, 0, 0, X, 1, D, 0, 0,
             bo, nullptr, 0, 0, 1.0f, (int)BN, D, D, 1, 1);

        // FF block
        layernorm_rows<<<(unsigned)BN, 256, 0, stream>>>(X, p_ln2_g + (long)l * D,
                                                         p_ln2_b + (long)l * D, Ybf);
        gemm(stream, Ybf, D, 1, 0, 0,
             p_ff1_w + (long)l * D * FF, 1, FF, 1, 0, 0,
             FFH, 3, FF, 0, 0,
             p_ff1_b + (long)l * FF, nullptr, 0, 0, 1.0f, (int)BN, FF, D, 1, 1);
        gemm(stream, FFH, FF, 1, 0, 0,
             p_ff2_w + (long)l * FF * D, 1, D, 1, 0, 0,
             X, 1, D, 0, 0,
             p_ff2_b + (long)l * D, nullptr, 0, 0, 1.0f, (int)BN, D, FF, 1, 1);
    }

    // ---- final pooling attention ----
    cvt_f32_bf16<<<(unsigned)((BN * D + 255) / 256), 256, 0, stream>>>(X, Ybf, BN * (long)D);
    gemm(stream, Ybf, (long)NTOK * D, 1, 0, 0, q_w, 1, D, 1, 0, 0,
         Qsp, 0, D, 0, 0, q_b, nullptr, 0, 0, 1.0f, B, D, D, 1, 1);
    gemm(stream, Ybf + D, D, 1, 0, (long)NTOK * D, k_w, 1, D, 1, 0, 0,
         KKm, 0, D, 0, (long)Kk * D, k_b, nullptr, 0, 0, 1.0f, Kk, D, D, B, 1);
    gemm(stream, Ybf + D, D, 1, 0, (long)NTOK * D, v_w, 1, D, 1, 0, 0,
         VVm, 0, D, 0, (long)Kk * D, v_b, nullptr, 0, 0, 1.0f, Kk, D, D, B, 1);

    long totS = (long)B * Kk;
    pool_scores<<<(unsigned)((totS + 255) / 256), 256, 0, stream>>>(Qsp, KKm, score, Kk, D,
                                                                    1.0f / 16.0f, totS);
    pool_softmax<<<B, 256, 0, stream>>>(score, Kk);
    long totP = (long)B * D;
    pool_weighted<<<(unsigned)((totP + 255) / 256), 256, 0, stream>>>(score, VVm, pool, Kk, D, totP);

    cvt_f32_bf16<<<(unsigned)((totP + 255) / 256), 256, 0, stream>>>(pool, poolb, totP);
    gemm(stream, poolb, D, 1, 0, 0, o_w, 1, D, 1, 0, 0,
         (float*)d_out, 0, D, 0, 0, o_b, nullptr, 0, 0, 1.0f, B, D, D, 1, 1);
}
